// LSTM_88802743812258
// MI455X (gfx1250) — compile-verified, matrix operand order checked by reading, not run on hardware
//
#include <hip/hip_runtime.h>
#include <hip/hip_bf16.h>
#include <stdint.h>
#include <stddef.h>

// Problem constants (match reference)
#define B_    512
#define T_IN_ 128
#define T_OUT_ 64
#define D_    64
#define H_    512
#define G_    2048   // 4*H

typedef __attribute__((ext_vector_type(16))) __bf16 v16bf;
typedef __attribute__((ext_vector_type(8)))  __bf16 v8bf;
typedef __attribute__((ext_vector_type(8)))  float  v8f;
typedef __attribute__((ext_vector_type(8)))  unsigned short ushort8;

// ---------- helpers ----------
__device__ __forceinline__ unsigned short f32_to_bf16_rne(float f) {
  union { float f; uint32_t u; } v; v.f = f;
  uint32_t u = v.u;
  if ((u & 0x7FFFFFFFu) > 0x7F800000u)            // NaN: keep quiet NaN
    return (unsigned short)((u >> 16) | 0x0040u);
  return (unsigned short)((u + 0x7FFFu + ((u >> 16) & 1u)) >> 16);
}

// Load one 16-element (per-lane) WMMA fragment half-pair:
// elements 0..7 at p[0..7], elements 8..15 at p[16..23] (ISA 16-bit 16x32 layout).
__device__ __forceinline__ v16bf load_frag(const unsigned short* __restrict__ p) {
  ushort8 lo = *(const ushort8*)(p);
  ushort8 hi = *(const ushort8*)(p + 16);
  v8bf blo = __builtin_bit_cast(v8bf, lo);
  v8bf bhi = __builtin_bit_cast(v8bf, hi);
  return __builtin_shufflevector(blo, bhi,
      0,1,2,3,4,5,6,7,8,9,10,11,12,13,14,15);
}

#define WMMA_BF16(a, b, c) \
  __builtin_amdgcn_wmma_f32_16x16x32_bf16(false, (a), false, (b), (short)0, (c), false, false)

// ---------- GEMM: C(MxN,f32) = A(MxK bf16,row,lda) * W(NxK bf16,row,ldw)^T ----------
// One wave computes a 32x64 strip of C (2 A-fragments x 4 B-fragments -> 8 WMMAs
// per 32-wide K chunk; ~21 FLOP/byte from cache). Requires M%32==0, N%64==0,
// K%32==0, rows 16B-aligned.
__global__ __launch_bounds__(256)
void gemm_bf16_wmma(const unsigned short* __restrict__ A, int lda,
                    const unsigned short* __restrict__ W, int ldw,
                    float* __restrict__ C, int ldc,
                    int M, int N, int K) {
  const int lane = threadIdx.x & 31;
  const int wave = blockIdx.x * (blockDim.x >> 5) + (threadIdx.x >> 5);
  const int tiles_n = N >> 6;
  const int tm = wave / tiles_n;
  const int tn = wave - tm * tiles_n;
  const int m0 = tm << 5;
  const int n0 = tn << 6;
  if (m0 >= M) return;

  const int lrow = lane & 15;
  const int koff = (lane >> 4) << 3;   // upper half-wave handles K+8 / K+24

  const unsigned short* a0 = A + (size_t)(m0 + lrow) * lda + koff;
  const unsigned short* a1 = a0 + (size_t)16 * lda;
  const unsigned short* w0 = W + (size_t)(n0 + lrow) * ldw + koff;
  const unsigned short* w1 = w0 + (size_t)16 * ldw;
  const unsigned short* w2 = w0 + (size_t)32 * ldw;
  const unsigned short* w3 = w0 + (size_t)48 * ldw;

  v8f acc00 = {}; v8f acc01 = {}; v8f acc02 = {}; v8f acc03 = {};
  v8f acc10 = {}; v8f acc11 = {}; v8f acc12 = {}; v8f acc13 = {};

  for (int k = 0; k < K; k += 32) {
    v16bf af0 = load_frag(a0 + k);
    v16bf af1 = load_frag(a1 + k);
    v16bf b0  = load_frag(w0 + k);
    acc00 = WMMA_BF16(af0, b0, acc00);
    acc10 = WMMA_BF16(af1, b0, acc10);
    v16bf b1  = load_frag(w1 + k);
    acc01 = WMMA_BF16(af0, b1, acc01);
    acc11 = WMMA_BF16(af1, b1, acc11);
    v16bf b2  = load_frag(w2 + k);
    acc02 = WMMA_BF16(af0, b2, acc02);
    acc12 = WMMA_BF16(af1, b2, acc12);
    v16bf b3  = load_frag(w3 + k);
    acc03 = WMMA_BF16(af0, b3, acc03);
    acc13 = WMMA_BF16(af1, b3, acc13);
  }

  // C/D layout: VGPR r, lanes 0-15 -> M=r, N=lane; lanes 16-31 -> M=8+r, N=lane-16
  const int crow = m0 + ((lane >> 4) << 3);
  float* cp0 = C + (size_t)crow * ldc + n0 + lrow;
  float* cp1 = cp0 + (size_t)16 * ldc;
#pragma unroll
  for (int r = 0; r < 8; ++r) {
    cp0[(size_t)r * ldc +  0] = acc00[r];
    cp0[(size_t)r * ldc + 16] = acc01[r];
    cp0[(size_t)r * ldc + 32] = acc02[r];
    cp0[(size_t)r * ldc + 48] = acc03[r];
    cp1[(size_t)r * ldc +  0] = acc10[r];
    cp1[(size_t)r * ldc + 16] = acc11[r];
    cp1[(size_t)r * ldc + 32] = acc12[r];
    cp1[(size_t)r * ldc + 48] = acc13[r];
  }
}

// ---------- LSTM elementwise cell ----------
// gates: (B, 4H) f32 (no bias yet). c: (B,H) f32 in/out.
// Writes h (bf16) to hdst1[b*ld1+off1+j] and optionally hdst2.
__global__ void lstm_elem(const float* __restrict__ gates,
                          const float* __restrict__ bih,
                          const float* __restrict__ bhh,
                          float* __restrict__ c,
                          unsigned short* __restrict__ hdst1, int ld1, int off1,
                          unsigned short* __restrict__ hdst2, int ld2, int off2) {
  int idx = blockIdx.x * blockDim.x + threadIdx.x;
  if (idx >= B_ * H_) return;
  int b = idx >> 9;          // / H_
  int j = idx & (H_ - 1);
  const float* g = gates + (size_t)b * G_;
  float gi = g[j]          + bih[j]          + bhh[j];
  float gf = g[H_ + j]     + bih[H_ + j]     + bhh[H_ + j];
  float gg = g[2*H_ + j]   + bih[2*H_ + j]   + bhh[2*H_ + j];
  float go = g[3*H_ + j]   + bih[3*H_ + j]   + bhh[3*H_ + j];
  float si = 1.0f / (1.0f + expf(-gi));
  float sf = 1.0f / (1.0f + expf(-gf));
  float so = 1.0f / (1.0f + expf(-go));
  float cn = sf * c[idx] + si * tanhf(gg);
  c[idx] = cn;
  float h = so * tanhf(cn);
  unsigned short hb = f32_to_bf16_rne(h);
  hdst1[(size_t)b * ld1 + off1 + j] = hb;
  if (hdst2) hdst2[(size_t)b * ld2 + off2 + j] = hb;
}

// ---------- misc small kernels ----------
// x part of xh0 <- bf16(data_in[:, t, :])
__global__ void load_x(const float* __restrict__ data_in, int t,
                       unsigned short* __restrict__ xh0) {
  int idx = blockIdx.x * blockDim.x + threadIdx.x;
  if (idx >= B_ * D_) return;
  int b = idx >> 6;
  int j = idx & (D_ - 1);
  xh0[(size_t)b * (D_ + H_) + j] =
      f32_to_bf16_rne(data_in[((size_t)b * T_IN_ + t) * D_ + j]);
}

// dst(2048 x (din+H)) bf16 = [Wih(2048 x din) | Whh(2048 x H)]
__global__ void build_w(const float* __restrict__ Wih,
                        const float* __restrict__ Whh,
                        int din, unsigned short* __restrict__ dst) {
  int ldw = din + H_;
  int total = G_ * ldw;
  for (int idx = blockIdx.x * blockDim.x + threadIdx.x; idx < total;
       idx += gridDim.x * blockDim.x) {
    int r = idx / ldw;
    int ccol = idx - r * ldw;
    float v = (ccol < din) ? Wih[(size_t)r * din + ccol]
                           : Whh[(size_t)r * H_ + (ccol - din)];
    dst[idx] = f32_to_bf16_rne(v);
  }
}

__global__ void convert_f32_bf16(const float* __restrict__ src,
                                 unsigned short* __restrict__ dst, int n) {
  for (int i = blockIdx.x * blockDim.x + threadIdx.x; i < n;
       i += gridDim.x * blockDim.x)
    dst[i] = f32_to_bf16_rne(src[i]);
}

__global__ void zero_f32(float* __restrict__ p, int n) {
  for (int i = blockIdx.x * blockDim.x + threadIdx.x; i < n;
       i += gridDim.x * blockDim.x) p[i] = 0.0f;
}

__global__ void zero_u16(unsigned short* __restrict__ p, int n) {
  for (int i = blockIdx.x * blockDim.x + threadIdx.x; i < n;
       i += gridDim.x * blockDim.x) p[i] = 0;
}

// proj (B x D) f32 + ba -> d_out column 0, and next-step x (bf16) in xh0
__global__ void proj_epi(const float* __restrict__ proj,
                         const float* __restrict__ ba,
                         unsigned short* __restrict__ xh0,
                         float* __restrict__ out, int t) {
  int idx = blockIdx.x * blockDim.x + threadIdx.x;
  if (idx >= B_ * D_) return;
  int b = idx >> 6;
  int j = idx & (D_ - 1);
  float v = proj[(size_t)b * D_ + j] + ba[j];
  xh0[(size_t)b * (D_ + H_) + j] = f32_to_bf16_rne(v);
  if (j == 0) out[(size_t)b * T_OUT_ + t] = v;
}

// ---------- host orchestration ----------
extern "C" void kernel_launch(void* const* d_in, const int* in_sizes, int n_in,
                              void* d_out, int out_size, void* d_ws, size_t ws_size,
                              hipStream_t stream) {
  const float* data_in = (const float*)d_in[0];
  // d_in[1] = true_out (unused)
  const float* e_Wih0 = (const float*)d_in[2];
  const float* e_Whh0 = (const float*)d_in[3];
  const float* e_bih0 = (const float*)d_in[4];
  const float* e_bhh0 = (const float*)d_in[5];
  const float* e_Wih1 = (const float*)d_in[6];
  const float* e_Whh1 = (const float*)d_in[7];
  const float* e_bih1 = (const float*)d_in[8];
  const float* e_bhh1 = (const float*)d_in[9];
  const float* d_Wih0 = (const float*)d_in[10];
  const float* d_Whh0 = (const float*)d_in[11];
  const float* d_bih0 = (const float*)d_in[12];
  const float* d_bhh0 = (const float*)d_in[13];
  const float* d_Wih1 = (const float*)d_in[14];
  const float* d_Whh1 = (const float*)d_in[15];
  const float* d_bih1 = (const float*)d_in[16];
  const float* d_bhh1 = (const float*)d_in[17];
  const float* Wa     = (const float*)d_in[18];
  const float* ba     = (const float*)d_in[19];
  float* out = (float*)d_out;

  // bump allocator over workspace (all chunks 256B aligned)
  uintptr_t p = (uintptr_t)d_ws;
  p = (p + 255) & ~(uintptr_t)255;
  auto alloc = [&](size_t bytes) {
    void* r = (void*)p;
    p += (bytes + 255) & ~(size_t)255;
    return r;
  };
  const int K0 = D_ + H_;    // 576
  const int K1 = H_ + H_;    // 1024
  unsigned short* wc0e = (unsigned short*)alloc((size_t)G_ * K0 * 2);
  unsigned short* wc1e = (unsigned short*)alloc((size_t)G_ * K1 * 2);
  unsigned short* wc0d = (unsigned short*)alloc((size_t)G_ * K0 * 2);
  unsigned short* wc1d = (unsigned short*)alloc((size_t)G_ * K1 * 2);
  unsigned short* wa_bf = (unsigned short*)alloc((size_t)D_ * H_ * 2);
  unsigned short* xh0  = (unsigned short*)alloc((size_t)B_ * K0 * 2);
  unsigned short* xh1  = (unsigned short*)alloc((size_t)B_ * K1 * 2);
  float* c0    = (float*)alloc((size_t)B_ * H_ * 4);
  float* c1    = (float*)alloc((size_t)B_ * H_ * 4);
  float* gates = (float*)alloc((size_t)B_ * G_ * 4);
  float* proj  = (float*)alloc((size_t)B_ * D_ * 4);

  hipStream_t s = stream;

  // ---- weight prep (bf16, [Wih|Whh] concat along K) ----
  build_w<<<1024, 256, 0, s>>>(e_Wih0, e_Whh0, D_, wc0e);
  build_w<<<2048, 256, 0, s>>>(e_Wih1, e_Whh1, H_, wc1e);
  build_w<<<1024, 256, 0, s>>>(d_Wih0, d_Whh0, D_, wc0d);
  build_w<<<2048, 256, 0, s>>>(d_Wih1, d_Whh1, H_, wc1d);
  convert_f32_bf16<<<128, 256, 0, s>>>(Wa, wa_bf, D_ * H_);

  // ---- init states (h=0 inside xh buffers, c=0) ----
  zero_u16<<<576, 256, 0, s>>>(xh0, B_ * K0);
  zero_u16<<<1024, 256, 0, s>>>(xh1, B_ * K1);
  zero_f32<<<1024, 256, 0, s>>>(c0, B_ * H_);
  zero_f32<<<1024, 256, 0, s>>>(c1, B_ * H_);

  // 32x64 per wave, 8 waves per block
  const int gate_blocks = (B_ / 32) * (G_ / 64) / 8;   // 64 blocks
  const int elem_blocks = (B_ * H_ + 255) / 256;       // 1024
  const int xd_blocks   = (B_ * D_ + 255) / 256;       // 128

  // ---- encoder: 128 steps, 2 layers ----
  for (int t = 0; t < T_IN_; ++t) {
    load_x<<<xd_blocks, 256, 0, s>>>(data_in, t, xh0);
    gemm_bf16_wmma<<<gate_blocks, 256, 0, s>>>(xh0, K0, wc0e, K0,
                                               gates, G_, B_, G_, K0);
    lstm_elem<<<elem_blocks, 256, 0, s>>>(gates, e_bih0, e_bhh0, c0,
                                          xh0, K0, D_,     // h0 into xh0[:,64:]
                                          xh1, K1, 0);     // h0 into xh1[:,0:512]
    gemm_bf16_wmma<<<gate_blocks, 256, 0, s>>>(xh1, K1, wc1e, K1,
                                               gates, G_, B_, G_, K1);
    lstm_elem<<<elem_blocks, 256, 0, s>>>(gates, e_bih1, e_bhh1, c1,
                                          xh1, K1, H_,     // h1 into xh1[:,512:]
                                          nullptr, 0, 0);
  }

  // ---- decoder: x0 = data_in[:, -1, :]; h/c carried over ----
  load_x<<<xd_blocks, 256, 0, s>>>(data_in, T_IN_ - 1, xh0);
  const int proj_blocks = ((B_ / 32) * (D_ / 64) + 7) / 8;   // 2 blocks
  for (int t = 0; t < T_OUT_; ++t) {
    gemm_bf16_wmma<<<gate_blocks, 256, 0, s>>>(xh0, K0, wc0d, K0,
                                               gates, G_, B_, G_, K0);
    lstm_elem<<<elem_blocks, 256, 0, s>>>(gates, d_bih0, d_bhh0, c0,
                                          xh0, K0, D_,
                                          xh1, K1, 0);
    gemm_bf16_wmma<<<gate_blocks, 256, 0, s>>>(xh1, K1, wc1d, K1,
                                               gates, G_, B_, G_, K1);
    lstm_elem<<<elem_blocks, 256, 0, s>>>(gates, d_bih1, d_bhh1, c1,
                                          xh1, K1, H_,
                                          nullptr, 0, 0);
    // projection: h1 view = xh1[:, 512:1024] (lda = 1024)
    gemm_bf16_wmma<<<proj_blocks, 256, 0, s>>>(xh1 + H_, K1, wa_bf, H_,
                                               proj, D_, B_, D_, H_);
    proj_epi<<<xd_blocks, 256, 0, s>>>(proj, ba, xh0, out, t);
  }
}
